// MultiHeadAttention_61744449847642
// MI455X (gfx1250) — matrix operand order repaired and checked by reading, not compiled, not run on hardware
//
#include <hip/hip_runtime.h>

// ---------------------------------------------------------------------------
// MI455X (gfx1250) multi-head attention forward.
// bf16 WMMA (v_wmma_f32_16x16x32_bf16) for all GEMMs + flash attention,
// f32 accumulation, online softmax in f32 VALU, causal mask computed inline.
// K tiles staged into LDS via the Tensor Data Mover (tensor_load_to_lds).
// ---------------------------------------------------------------------------

typedef __attribute__((ext_vector_type(16))) __bf16 v16bf;
typedef __attribute__((ext_vector_type(8)))  float  v8f;
typedef __attribute__((ext_vector_type(4)))  unsigned int v4u;
typedef __attribute__((ext_vector_type(8)))  int v8i;
typedef __attribute__((ext_vector_type(4)))  int v4i;
typedef unsigned short bf16_t;   // raw bf16 bits

union F8      { v8f v;    float f[8];  };
union BF16x16 { uint4 q[2]; v16bf v;   };

__device__ __forceinline__ bf16_t f2bf(float x) {
  unsigned u = __float_as_uint(x);
  u += 0x7fffu + ((u >> 16) & 1u);   // round-to-nearest-even
  return (bf16_t)(u >> 16);
}

// Generic pointer -> 32-bit LDS byte offset (addrspacecast to as(3), ptrtoint)
__device__ __forceinline__ unsigned lds_offset(const void* p) {
  return (unsigned)(size_t)(__attribute__((address_space(3))) const void*)p;
}

// ---------------------------------------------------------------------------
// TDM: async 2-D tile load Global -> LDS (Tensor DMA Descriptor per ISA 8.3/8.4)
// data_size = 2 bytes; optional LDS row padding.
//  pad_interval_code: 0=2,1=4,2=8,3=16,4=32,... DWORDs between pads
//  pad_amount_code:   0=1,1=2,...               DWORDs of pad inserted
// ---------------------------------------------------------------------------
__device__ __forceinline__ void tdm_load_2d_bf16(
    unsigned lds_addr, const void* gaddr,
    unsigned tensor_d0, unsigned tensor_d1,
    unsigned tile_d0, unsigned tile_d1, unsigned stride0,
    unsigned pad_enable, unsigned pad_interval_code, unsigned pad_amount_code)
{
  const unsigned long long ga = (unsigned long long)(size_t)gaddr;
  v4u g0;
  g0.x = 1u;                                           // count=1 (valid user D#)
  g0.y = lds_addr;                                     // lds_addr   [63:32]
  g0.z = (unsigned)(ga & 0xffffffffu);                 // global_addr[95:64]
  g0.w = (unsigned)((ga >> 32) & 0x01ffffffu)          // global_addr[120:96]
       | 0x80000000u;                                  // type=2 ("image") [127:126]
  v8i g1;
  g1[0] = (int)((1u << 16)                             // data_size = 2 bytes
              | (pad_enable << 20)
              | (pad_interval_code << 22)
              | (pad_amount_code << 25));
  g1[1] = (int)(tensor_d0 << 16);                      // tensor_dim0[15:0] @ [63:48]
  g1[2] = (int)((tensor_d0 >> 16) | (tensor_d1 << 16));// dim0 hi | dim1 lo
  g1[3] = (int)((tensor_d1 >> 16) | (tile_d0 << 16));  // dim1 hi | tile_dim0
  g1[4] = (int)(tile_d1 & 0xffffu);                    // tile_dim1 (tile_dim2=0)
  g1[5] = (int)stride0;                                // tensor_dim0_stride lo32
  g1[6] = 0;                                           // stride0 hi | dim1_stride lo
  g1[7] = 0;
  const v4i z4 = {0, 0, 0, 0};                         // 2-D tensor: groups 2/3 unused
  const v8i z8 = {0, 0, 0, 0, 0, 0, 0, 0};
  __builtin_amdgcn_tensor_load_to_lds(g0, g1, z4, z4, z8, 0);
}

// A-matrix fragment, 16x32 bf16 (ISA 7.12.2): lanes 0-15 hold row m=lane,
// K = {k0..k0+7, k0+16..k0+23}; lanes 16-31 hold row m=lane-16,
// K = {k0+8..k0+15, k0+24..k0+31}. Two 16B loads per lane.
__device__ __forceinline__ v16bf load_frag_a(const bf16_t* base, int ld, int k0) {
  const int lane = threadIdx.x & 31;
  const int r = lane & 15, hi = lane >> 4;
  const bf16_t* p = base + (size_t)r * ld + k0 + hi * 8;
  BF16x16 u;
  u.q[0] = *(const uint4*)(p);
  u.q[1] = *(const uint4*)(p + 16);
  return u.v;
}

// B-matrix fragment, 32x16 bf16: lanes 0-15 hold column n=lane with
// K = {k0..k0+7, k0+16..k0+23}; lanes 16-31 hold column n=lane-16, K = {k0+8..k0+15, k0+24..k0+31}.
__device__ __forceinline__ v16bf load_frag_b(const bf16_t* base, int ld, int k0) {
  const int lane = threadIdx.x & 31;
  const int r = lane & 15, hi = lane >> 4;
  const bf16_t* p = base + (size_t)r * ld + k0 + hi * 8;
  BF16x16 u;
  u.q[0] = *(const uint4*)(p);
  u.q[1] = *(const uint4*)(p + 16);
  return u.v;
}

#define WMMA_BF16(a, b, c) \
  __builtin_amdgcn_wmma_f32_16x16x32_bf16(false, (a), false, (b), (short)0, (c), false, false)

// ---------------------------------------------------------------------------
// fp32 -> bf16 cast
// ---------------------------------------------------------------------------
__global__ void cast_f32_to_bf16(const float* __restrict__ in,
                                 bf16_t* __restrict__ out, int n) {
  int i = blockIdx.x * blockDim.x + threadIdx.x;
  if (i < n) out[i] = f2bf(in[i]);
}

// ---------------------------------------------------------------------------
// Y[M,N] = A[M,K](bf16) * W[N,K]^T(bf16) + bias  (reference einsum 'bsd,ed->bse')
// Block: 256 threads = 8 waves; block tile 128x128; wave tile 32x64.
// ---------------------------------------------------------------------------
__global__ __launch_bounds__(256, 2) void gemm_bf16_kernel(
    const bf16_t* __restrict__ A, const bf16_t* __restrict__ Wt,
    const float* __restrict__ bias, float* __restrict__ Y,
    int M, int N, int K)
{
  const int lane = threadIdx.x & 31;
  const int wave = threadIdx.x >> 5;
  const int wm = wave & 3, wn = wave >> 2;
  const int m0 = blockIdx.x * 128 + wm * 32;
  const int n0 = blockIdx.y * 128 + wn * 64;

  const v8f vzero = {0.f, 0.f, 0.f, 0.f, 0.f, 0.f, 0.f, 0.f};
  v8f acc[2][4];
#pragma unroll
  for (int i = 0; i < 2; ++i)
#pragma unroll
    for (int t = 0; t < 4; ++t) acc[i][t] = vzero;

  const bf16_t* a0 = A + (size_t)m0 * K;
  const bf16_t* a1 = A + (size_t)(m0 + 16) * K;
  const bf16_t* bbase = Wt + (size_t)n0 * K;

  for (int k0 = 0; k0 < K; k0 += 32) {
    // speculative prefetch of the next K-slice (global_prefetch_b8)
    __builtin_prefetch(a0 + (size_t)(lane & 15) * K + k0 + 64, 0, 1);
    __builtin_prefetch(bbase + (size_t)(lane & 15) * K + k0 + 64, 0, 1);

    v16bf af0 = load_frag_a(a0, K, k0);
    v16bf af1 = load_frag_a(a1, K, k0);
#pragma unroll
    for (int t = 0; t < 4; ++t) {
      v16bf bfm = load_frag_b(bbase + (size_t)t * 16 * K, K, k0);
      acc[0][t] = WMMA_BF16(af0, bfm, acc[0][t]);
      acc[1][t] = WMMA_BF16(af1, bfm, acc[1][t]);
    }
  }

  const int cl = lane & 15, hi = lane >> 4;
#pragma unroll
  for (int i = 0; i < 2; ++i)
#pragma unroll
    for (int t = 0; t < 4; ++t) {
      F8 u; u.v = acc[i][t];
      const int col = n0 + t * 16 + cl;
      const float bb = bias ? bias[col] : 0.0f;
#pragma unroll
      for (int r = 0; r < 8; ++r) {
        const int row = m0 + i * 16 + r + 8 * hi;
        Y[(size_t)row * N + col] = u.f[r] + bb;
      }
    }
}

// ---------------------------------------------------------------------------
// RoPE + repack: X fp32 [B,S,H*64] -> Y bf16 [B*H, S, 64]
// pairs (d, d+32) within each 64-wide head; do_rope=0 -> plain repack (V path)
// ---------------------------------------------------------------------------
__global__ void rope_pack_kernel(const float* __restrict__ X,
                                 const float* __restrict__ cosb,
                                 const float* __restrict__ sinb,
                                 bf16_t* __restrict__ Y, int do_rope)
{
  const int idx = blockIdx.x * blockDim.x + threadIdx.x;
  const int total = 2 * 2048 * 16 * 32;
  if (idx >= total) return;
  const int d = idx & 31;
  const int h = (idx >> 5) & 15;
  const int s = (idx >> 9) & 2047;
  const int b = idx >> 20;

  const float* xr = X + ((size_t)(b * 2048 + s)) * 1024 + h * 64;
  const float x1 = xr[d], x2 = xr[d + 32];
  float y1 = x1, y2 = x2;
  if (do_rope) {
    const float c = cosb[s * 32 + d], sn = sinb[s * 32 + d];
    y1 = x1 * c - x2 * sn;
    y2 = x1 * sn + x2 * c;
  }
  bf16_t* yr = Y + ((size_t)(b * 16 + h) * 2048 + s) * 64;
  yr[d]      = f2bf(y1);
  yr[d + 32] = f2bf(y2);
}

// ---------------------------------------------------------------------------
// Flash attention (causal), hd=64, kv-tile = 64.
// 128 threads = 4 waves; each wave owns 16 query rows (64 rows/block).
// K tile DMA'd into LDS by the TDM (wave 0) with native row padding; V tile
// transposed into LDS by all threads (overlapped with the TDM transfer);
// P staged per-wave in LDS as bf16.
// Output: bf16 [B, S, H*64] ready for the output projection GEMM.
// ---------------------------------------------------------------------------
__global__ __launch_bounds__(128, 2) void flash_attn_kernel(
    const bf16_t* __restrict__ Q, const bf16_t* __restrict__ Kc,
    const bf16_t* __restrict__ Vc, bf16_t* __restrict__ Og, int S)
{
  __shared__ __align__(16) bf16_t Ks[64][72];
  __shared__ __align__(16) bf16_t Vt[64][72];
  __shared__ __align__(16) bf16_t Ps[4][16][72];

  const int tid  = threadIdx.x;
  const int lane = tid & 31, wave = tid >> 5;
  const int cl = lane & 15, hi = lane >> 4;
  const int qblk = blockIdx.x;
  const int bh   = blockIdx.y;
  const int b = bh >> 4, h = bh & 15;
  const int qrow0 = qblk * 64 + wave * 16;

  const bf16_t* Qbase = Q + ((size_t)bh * S + qrow0) * 64;
  const v16bf qa0 = load_frag_a(Qbase, 64, 0);
  const v16bf qa1 = load_frag_a(Qbase, 64, 32);

  const unsigned ksLds = lds_offset(&Ks[0][0]);

  const v8f vzero = {0.f, 0.f, 0.f, 0.f, 0.f, 0.f, 0.f, 0.f};
  F8 o[4];
#pragma unroll
  for (int t = 0; t < 4; ++t) o[t].v = vzero;
  float mrow[8], lrow[8];
#pragma unroll
  for (int r = 0; r < 8; ++r) { mrow[r] = -3.0e38f; lrow[r] = 0.f; }

  const float scale = 0.125f;  // hd^-0.5, hd=64
  const int kvend = qblk * 64 + 64;

  for (int kv = 0; kv < kvend; kv += 64) {
    const bf16_t* Kg = Kc + ((size_t)bh * S + kv) * 64;
    const bf16_t* Vg = Vc + ((size_t)bh * S + kv) * 64;

    // K tile: TDM async DMA global->LDS, 64x64 bf16 tile, LDS rows padded
    // 128B data + 16B pad (pad_interval 32 DWORDs = code 4, pad 4 DWORDs = code 3)
    if (wave == 0) {
      tdm_load_2d_bf16(ksLds, Kg,
                       /*tensor_d0=*/64u, /*tensor_d1=*/(unsigned)S,
                       /*tile_d0=*/64u, /*tile_d1=*/64u, /*stride0=*/64u,
                       /*pad_enable=*/1u, /*pad_interval=*/4u, /*pad_amount=*/3u);
    }
    // V tile: transpose into LDS (coalesced global reads), overlapped with TDM
    for (int i = tid; i < 64 * 64; i += 128) {
      const int row = i >> 6, c = i & 63;
      Vt[c][row] = Vg[row * 64 + c];
    }
    if (wave == 0) __builtin_amdgcn_s_wait_tensorcnt(0);
    __syncthreads();

    // S = Q K^T : 4 column tiles of 16, two K-steps over hd=64
    F8 s[4];
#pragma unroll
    for (int t = 0; t < 4; ++t) {
      v8f z = vzero;
      z = WMMA_BF16(qa0, load_frag_b(&Ks[t * 16][0], 72, 0),  z);
      z = WMMA_BF16(qa1, load_frag_b(&Ks[t * 16][0], 72, 32), z);
      s[t].v = z;
    }

    // online softmax (per C-layout row r / r+8; stats replicated in 16-lane half)
#pragma unroll
    for (int r = 0; r < 8; ++r) {
      const int row = qrow0 + r + 8 * hi;
      float mx = -3.0e38f;
#pragma unroll
      for (int t = 0; t < 4; ++t) {
        const int col = kv + t * 16 + cl;
        const float v = s[t].f[r] * scale + (col <= row ? 0.f : -1.0e9f);
        s[t].f[r] = v;
        mx = fmaxf(mx, v);
      }
#pragma unroll
      for (int d = 1; d < 16; d <<= 1) mx = fmaxf(mx, __shfl_xor(mx, d, 32));
      const float nm = fmaxf(mrow[r], mx);
      const float alpha = __expf(mrow[r] - nm);
      mrow[r] = nm;
      lrow[r] *= alpha;
      float ps = 0.f;
#pragma unroll
      for (int t = 0; t < 4; ++t) {
        const float p = __expf(s[t].f[r] - nm);
        s[t].f[r] = p;
        ps += p;
        o[t].f[r] *= alpha;
      }
#pragma unroll
      for (int d = 1; d < 16; d <<= 1) ps += __shfl_xor(ps, d, 32);
      lrow[r] += ps;
    }

    // P -> per-wave LDS as bf16 (same-wave DS ordering guarantees visibility)
#pragma unroll
    for (int t = 0; t < 4; ++t)
#pragma unroll
      for (int r = 0; r < 8; ++r)
        Ps[wave][r + 8 * hi][t * 16 + cl] = f2bf(s[t].f[r]);

    // O += P V : K-dim = 64 keys, two steps of 32
#pragma unroll
    for (int j = 0; j < 2; ++j) {
      const v16bf pa = load_frag_a(&Ps[wave][0][0], 72, j * 32);
#pragma unroll
      for (int t = 0; t < 4; ++t) {
        const v16bf vb = load_frag_b(&Vt[t * 16][0], 72, j * 32);
        o[t].v = WMMA_BF16(pa, vb, o[t].v);
      }
    }
    __syncthreads();
  }

  // normalize and store as bf16 [B, S, H*64]
#pragma unroll
  for (int t = 0; t < 4; ++t)
#pragma unroll
    for (int r = 0; r < 8; ++r) {
      const int row = qrow0 + r + 8 * hi;
      const float val = o[t].f[r] / lrow[r];
      Og[((size_t)b * S + row) * 1024 + h * 64 + t * 16 + cl] = f2bf(val);
    }
}

// ---------------------------------------------------------------------------
// host-side launch
// ---------------------------------------------------------------------------
extern "C" void kernel_launch(void* const* d_in, const int* in_sizes, int n_in,
                              void* d_out, int out_size, void* d_ws, size_t ws_size,
                              hipStream_t stream) {
  const float* x    = (const float*)d_in[0];
  const float* cosb = (const float*)d_in[1];
  const float* sinb = (const float*)d_in[2];
  // d_in[3] = mask : causal mask computed inline, not read
  const float* Wq = (const float*)d_in[4];
  const float* bq = (const float*)d_in[5];
  const float* Wk = (const float*)d_in[6];
  const float* bk = (const float*)d_in[7];
  const float* Wv = (const float*)d_in[8];
  const float* bv = (const float*)d_in[9];
  const float* Wo = (const float*)d_in[10];
  const float* bo = (const float*)d_in[11];
  (void)in_sizes; (void)n_in; (void)out_size; (void)ws_size;

  const int S = 2048, D = 1024;
  const size_t NX = (size_t)2 * S * D;  // 4M elements
  const size_t NW = (size_t)D * D;      // 1M elements

  char* w = (char*)d_ws;
  bf16_t* xb  = (bf16_t*)w; w += NX * 2;
  bf16_t* wqb = (bf16_t*)w; w += NW * 2;
  bf16_t* wkb = (bf16_t*)w; w += NW * 2;
  bf16_t* wvb = (bf16_t*)w; w += NW * 2;
  bf16_t* wob = (bf16_t*)w; w += NW * 2;
  float*  tmp = (float*)w;  w += NX * 4;
  bf16_t* Qb  = (bf16_t*)w; w += NX * 2;
  bf16_t* Kb  = (bf16_t*)w; w += NX * 2;
  bf16_t* Vb  = (bf16_t*)w; w += NX * 2;
  bf16_t* Ab  = (bf16_t*)w; w += NX * 2;   // attention out, bf16 [B,S,D]

  // casts
  cast_f32_to_bf16<<<(int)((NX + 255) / 256), 256, 0, stream>>>(x,  xb,  (int)NX);
  cast_f32_to_bf16<<<(int)((NW + 255) / 256), 256, 0, stream>>>(Wq, wqb, (int)NW);
  cast_f32_to_bf16<<<(int)((NW + 255) / 256), 256, 0, stream>>>(Wk, wkb, (int)NW);
  cast_f32_to_bf16<<<(int)((NW + 255) / 256), 256, 0, stream>>>(Wv, wvb, (int)NW);
  cast_f32_to_bf16<<<(int)((NW + 255) / 256), 256, 0, stream>>>(Wo, wob, (int)NW);

  const dim3 ggrid(32, 8), gblk(256);
  const int ropeGrid = (2 * 2048 * 16 * 32 + 255) / 256;

  // Q = rope(x Wq^T + bq)
  gemm_bf16_kernel<<<ggrid, gblk, 0, stream>>>(xb, wqb, bq, tmp, 4096, 1024, 1024);
  rope_pack_kernel<<<ropeGrid, 256, 0, stream>>>(tmp, cosb, sinb, Qb, 1);
  // K = rope(x Wk^T + bk)
  gemm_bf16_kernel<<<ggrid, gblk, 0, stream>>>(xb, wkb, bk, tmp, 4096, 1024, 1024);
  rope_pack_kernel<<<ropeGrid, 256, 0, stream>>>(tmp, cosb, sinb, Kb, 1);
  // V = x Wv^T + bv
  gemm_bf16_kernel<<<ggrid, gblk, 0, stream>>>(xb, wvb, bv, tmp, 4096, 1024, 1024);
  rope_pack_kernel<<<ropeGrid, 256, 0, stream>>>(tmp, cosb, sinb, Vb, 0);

  // attention
  flash_attn_kernel<<<dim3(32, 32), 128, 0, stream>>>(Qb, Kb, Vb, Ab, S);

  // out = attn Wo^T + bo  (fp32 to d_out)
  gemm_bf16_kernel<<<ggrid, gblk, 0, stream>>>(Ab, wob, bo, (float*)d_out, 4096, 1024, 1024);
}
